// IAF_74423193305610
// MI455X (gfx1250) — matrix operand order repaired and checked by reading, not compiled, not run on hardware
//
#include <hip/hip_runtime.h>
#include <hip/hip_bf16.h>

typedef __attribute__((ext_vector_type(16))) _Float16 v16h;
typedef __attribute__((ext_vector_type(8)))  _Float16 v8h;
typedef __attribute__((ext_vector_type(8)))  float    v8f;

#define B_    2048
#define D_    32
#define H_    512
#define ST    520   // LDS row stride in halves (padded: 1040B = 16B*65, breaks bank conflicts)

// ---------------------------------------------------------------------------
// Weight packing: apply MADE masks, convert to f16, lay out in WMMA B-fragment
// order. B fragment (16x16x32 f16 WMMA): lane L holds column n = nt*16 + (L&15),
// K-run kt*32 + (L>>4)*16 + [0..16). Packed so each lane reads 32 contiguous B.
// frag index = nt * (K/32) + kt  (kt innermost => streaming in the GEMM k-loop).
// ---------------------------------------------------------------------------
__global__ void pack_weights(const float* __restrict__ W, _Float16* __restrict__ dst,
                             int K, int N, int mode) {
  long idx = (long)blockIdx.x * blockDim.x + threadIdx.x;
  long per_step = (long)K * N;
  long total = 2 * per_step;            // STEPS = 2
  if (idx >= total) return;
  int s = (int)(idx / per_step);
  long r = idx % per_step;
  int k = (int)(r / N);
  int n = (int)(r % N);
  float w = W[(long)s * per_step + (long)k * N + n];
  bool keep;
  if (mode == 0)      keep = (k + 1) <  (n / 16 + 1);   // m_in : deg_in[k] <  deg_h[n]
  else if (mode == 1) keep = (k / 16) <= (n / 16);      // m_h  : deg_h[k]  <= deg_h[n]
  else                keep = (k / 16) <= n;             // m_out: deg_h[k]  <= deg_in[n]
  float v = keep ? w : 0.0f;

  int kt = k >> 5, kk = k & 31;
  int nt = n >> 4, c  = n & 15;
  int lane = ((kk >> 4) << 4) | c;
  int h    = kk & 15;
  int ntk  = K >> 5;
  long nfrag = (long)(N >> 4) * ntk;
  long frag  = (long)nt * ntk + kt;
  long di = (((long)s * nfrag + frag) * 32 + lane) * 16 + h;
  dst[di] = (_Float16)v;
}

// ---------------------------------------------------------------------------
// Fragment helpers (wave32).
// A fragment 16x32 f16: lane L -> row = L&15, kb = (L>>4)*16,
// holds K = kb+[0..16) (matches packed B k order): two 16B LDS loads.
// ---------------------------------------------------------------------------
__device__ __forceinline__ v16h load_afrag(const _Float16* buf, int lane, int kt) {
  int row = lane & 15;
  int kb  = (lane >> 4) << 4;
  const _Float16* p = buf + row * ST + kt * 32 + kb;
  v8h lo = *(const v8h*)(p);
  v8h hi = *(const v8h*)(p + 8);
  return __builtin_shufflevector(lo, hi, 0,1,2,3,4,5,6,7,8,9,10,11,12,13,14,15);
}

__device__ __forceinline__ v16h load_bfrag(const _Float16* w, long frag, int lane) {
  return *(const v16h*)(w + (frag * 32 + lane) * 16);
}

// Store the two f32 D-fragments (16 rows x 32 cols) to LDS as f16 row-major.
__device__ __forceinline__ void store_xfrags(_Float16* buf, const v8f x0, const v8f x1, int lane) {
  int c = lane & 15, g = lane >> 4;
  _Float16* p = buf + (g * 8) * ST + c;
#pragma unroll
  for (int e = 0; e < 8; ++e) {
    p[e * ST]      = (_Float16)x0[e];
    p[e * ST + 16] = (_Float16)x1[e];
  }
}

// One hidden layer: bufIn (16 x 32*KT f16) -> GEMM vs packed weights -> ReLU
// -> bufOut (16 x 512 f16). N = 512 (32 n-tiles).
template<int KT>
__device__ __forceinline__ void layer_to_lds(const _Float16* bufIn, _Float16* bufOut,
                                             const _Float16* __restrict__ w,
                                             const float* __restrict__ bias, int lane) {
  int c = lane & 15, g = lane >> 4;
  v16h A[KT];
#pragma unroll
  for (int kt = 0; kt < KT; ++kt) A[kt] = load_afrag(bufIn, lane, kt);

#pragma unroll 1
  for (int nt = 0; nt < 32; ++nt) {
    if (nt + 1 < 32)  // stream next weight tile toward L2/L0
      __builtin_prefetch(w + ((long)(nt + 1) * KT * 32 + lane) * 16, 0, 1);
    float bv = bias[nt * 16 + c];
    v8f acc = {bv, bv, bv, bv, bv, bv, bv, bv};
#pragma unroll
    for (int kt = 0; kt < KT; ++kt) {
      v16h Bf = load_bfrag(w, (long)nt * KT + kt, lane);
      acc = __builtin_amdgcn_wmma_f32_16x16x32_f16(false, A[kt], false, Bf,
                                                   (short)0, acc, false, false);
    }
    _Float16* op = bufOut + (g * 8) * ST + nt * 16 + c;
#pragma unroll
    for (int e = 0; e < 8; ++e) {
      float v = acc[e];
      v = v > 0.0f ? v : 0.0f;   // ReLU
      op[e * ST] = (_Float16)v;
    }
  }
}

// Output layer: (16 x 512) @ (512 x 32), no ReLU, results stay in registers.
__device__ __forceinline__ void layer_out(const _Float16* bufIn,
                                          const _Float16* __restrict__ w,
                                          const float* __restrict__ bias,
                                          int lane, v8f& o0, v8f& o1) {
  int c = lane & 15;
  v16h A[16];
#pragma unroll
  for (int kt = 0; kt < 16; ++kt) A[kt] = load_afrag(bufIn, lane, kt);
  v8f out[2];
#pragma unroll 1
  for (int nt = 0; nt < 2; ++nt) {
    float bv = bias[nt * 16 + c];
    v8f acc = {bv, bv, bv, bv, bv, bv, bv, bv};
#pragma unroll
    for (int kt = 0; kt < 16; ++kt) {
      v16h Bf = load_bfrag(w, (long)nt * 16 + kt, lane);
      acc = __builtin_amdgcn_wmma_f32_16x16x32_f16(false, A[kt], false, Bf,
                                                   (short)0, acc, false, false);
    }
    out[nt] = acc;
  }
  o0 = out[0]; o1 = out[1];
}

struct StepW {
  const _Float16 *w0, *w1, *w2, *w3;
  const float *b0, *b1, *b2, *b3;
};

__device__ void made_shift(const v8f x0, const v8f x1, v8f& s0, v8f& s1,
                           _Float16* bufA, _Float16* bufB, const StepW& sw, int lane) {
  store_xfrags(bufA, x0, x1, lane);
  layer_to_lds<1>(bufA, bufB, sw.w0, sw.b0, lane);   // 32  -> 512
  layer_to_lds<16>(bufB, bufA, sw.w1, sw.b1, lane);  // 512 -> 512
  layer_to_lds<16>(bufA, bufB, sw.w2, sw.b2, lane);  // 512 -> 512
  layer_out(bufB, sw.w3, sw.b3, lane, s0, s1);       // 512 -> 32
}

// x[..., ::-1] : swap the two 16-col tiles and mirror columns within a tile.
// Column lives in lane bits [3:0] => lane XOR 15 (stays within 16-lane group).
__device__ __forceinline__ void reverse2(v8f& x0, v8f& x1) {
  v8f n0, n1;
#pragma unroll
  for (int e = 0; e < 8; ++e) {
    n0[e] = __shfl_xor(x1[e], 15, 32);
    n1[e] = __shfl_xor(x0[e], 15, 32);
  }
  x0 = n0; x1 = n1;
}

// ---------------------------------------------------------------------------
// Main kernel: one wave (32 lanes) per 16 batch rows, runs the entire flow.
// ---------------------------------------------------------------------------
__global__ void __launch_bounds__(32)
iaf_main(const float* __restrict__ zmean, const float* __restrict__ zlogv,
         const float* __restrict__ eps,
         const float* __restrict__ b0, const float* __restrict__ b1,
         const float* __restrict__ b2, const float* __restrict__ b3,
         const _Float16* __restrict__ pw0, const _Float16* __restrict__ pw1,
         const _Float16* __restrict__ pw2, const _Float16* __restrict__ pw3,
         float* __restrict__ out_samples, float* __restrict__ out_logp) {
  __shared__ _Float16 bufA[16 * ST];
  __shared__ _Float16 bufB[16 * ST];

  int lane = threadIdx.x;
  int c = lane & 15, g = lane >> 4;
  int row0 = blockIdx.x * 16;

  // Load z in D-fragment layout: element e of tile t = (row 8g+e, col t*16+c).
  v8f zm[2], lv[2], ep[2], stdv[2], x0v, x1v;
#pragma unroll
  for (int t = 0; t < 2; ++t)
#pragma unroll
    for (int e = 0; e < 8; ++e) {
      int idx = (row0 + g * 8 + e) * D_ + t * 16 + c;
      zm[t][e] = zmean[idx];
      lv[t][e] = zlogv[idx];
      ep[t][e] = eps[idx];
    }
#pragma unroll
  for (int t = 0; t < 2; ++t)
#pragma unroll
    for (int e = 0; e < 8; ++e)
      stdv[t][e] = __expf(0.5f * lv[t][e]);
#pragma unroll
  for (int e = 0; e < 8; ++e) {
    x0v[e] = zm[0][e] + stdv[0][e] * ep[0][e];
    x1v[e] = zm[1][e] + stdv[1][e] * ep[1][e];
  }

  const StepW sw0 = { pw0,            pw1,              pw2,              pw3,
                      b0,             b1,               b2,               b3 };
  const StepW sw1 = { pw0 + 32 * 512, pw1 + 512 * 512,  pw2 + 512 * 512,  pw3 + 32 * 512,
                      b0 + 512,       b1 + 512,         b2 + 512,         b3 + 32 };

  v8f s0, s1;

  // ---- inverse pass: i = 1 then i = 0, reverse after i = 1 ----
  made_shift(x0v, x1v, s0, s1, bufA, bufB, sw1, lane);
#pragma unroll
  for (int e = 0; e < 8; ++e) { x0v[e] -= s0[e]; x1v[e] -= s1[e]; }
  reverse2(x0v, x1v);
  made_shift(x0v, x1v, s0, s1, bufA, bufB, sw0, lane);
#pragma unroll
  for (int e = 0; e < 8; ++e) { x0v[e] -= s0[e]; x1v[e] -= s1[e]; }

  // samples
#pragma unroll
  for (int e = 0; e < 8; ++e) {
    out_samples[(row0 + g * 8 + e) * D_ + c]      = x0v[e];
    out_samples[(row0 + g * 8 + e) * D_ + 16 + c] = x1v[e];
  }

  // ---- forward pass: 2 steps x 32 fixed-point scan iterations ----
  v8f y0 = x0v, y1 = x1v;
#pragma unroll 1
  for (int i = 0; i < 2; ++i) {
    const StepW* sw = i ? &sw1 : &sw0;
    v8f xin0 = y0, xin1 = y1;
    v8f yy0 = {0,0,0,0,0,0,0,0}, yy1 = {0,0,0,0,0,0,0,0};
#pragma unroll 1
    for (int it = 0; it < D_; ++it) {
      made_shift(yy0, yy1, s0, s1, bufA, bufB, *sw, lane);
#pragma unroll
      for (int e = 0; e < 8; ++e) { yy0[e] = xin0[e] + s0[e]; yy1[e] = xin1[e] + s1[e]; }
    }
    y0 = yy0; y1 = yy1;
    if (i == 0) reverse2(y0, y1);
  }

  // ---- logp: per-row reduction over 32 cols (16-lane xor reduction) ----
  float ssq[8], slv[8];
#pragma unroll
  for (int e = 0; e < 8; ++e) {
    float d0 = (y0[e] - zm[0][e]) / stdv[0][e];
    float d1 = (y1[e] - zm[1][e]) / stdv[1][e];
    ssq[e] = d0 * d0 + d1 * d1;
    slv[e] = lv[0][e] + lv[1][e];
  }
#pragma unroll
  for (int off = 1; off < 16; off <<= 1)
#pragma unroll
    for (int e = 0; e < 8; ++e) {
      ssq[e] += __shfl_xor(ssq[e], off, 32);
      slv[e] += __shfl_xor(slv[e], off, 32);
    }
  if (c < 8) {
    const float LOG2PI = 1.8378770664093454836f;
    float lp = 0.0f;
#pragma unroll
    for (int e = 0; e < 8; ++e) if (e == c) lp = -0.5f * ssq[e] - 0.5f * slv[e];
    lp -= 0.5f * (float)D_ * LOG2PI;
    out_logp[row0 + g * 8 + c] = lp;
  }
}

// ---------------------------------------------------------------------------
extern "C" void kernel_launch(void* const* d_in, const int* in_sizes, int n_in,
                              void* d_out, int out_size, void* d_ws, size_t ws_size,
                              hipStream_t stream) {
  const float* zmean = (const float*)d_in[0];
  const float* zlogv = (const float*)d_in[1];
  const float* eps   = (const float*)d_in[2];
  const float* W0    = (const float*)d_in[3];
  const float* b0    = (const float*)d_in[4];
  const float* W1    = (const float*)d_in[5];
  const float* b1    = (const float*)d_in[6];
  const float* W2    = (const float*)d_in[7];
  const float* b2    = (const float*)d_in[8];
  const float* W3    = (const float*)d_in[9];
  const float* b3    = (const float*)d_in[10];

  // Workspace: packed masked f16 weights, step-major, B-fragment layout.
  _Float16* pw0 = (_Float16*)d_ws;               // 2 * 32*512   = 32768 halves
  _Float16* pw1 = pw0 + 2 * 32 * 512;            // 2 * 512*512  = 524288
  _Float16* pw2 = pw1 + 2 * 512 * 512;           // 2 * 512*512
  _Float16* pw3 = pw2 + 2 * 512 * 512;           // 2 * 512*32   = 32768

  pack_weights<<<(2 * 32 * 512  + 255) / 256, 256, 0, stream>>>(W0, pw0,  32, 512, 0);
  pack_weights<<<(2 * 512 * 512 + 255) / 256, 256, 0, stream>>>(W1, pw1, 512, 512, 1);
  pack_weights<<<(2 * 512 * 512 + 255) / 256, 256, 0, stream>>>(W2, pw2, 512, 512, 1);
  pack_weights<<<(2 * 512 * 32  + 255) / 256, 256, 0, stream>>>(W3, pw3, 512,  32, 2);

  float* out_samples = (float*)d_out;
  float* out_logp    = out_samples + (long)B_ * D_;

  iaf_main<<<B_ / 16, 32, 0, stream>>>(zmean, zlogv, eps, b0, b1, b2, b3,
                                       pw0, pw1, pw2, pw3, out_samples, out_logp);
}
